// CooccurrenceMatrix_37074157699300
// MI455X (gfx1250) — matrix operand order repaired and checked by reading, not compiled, not run on hardware
//
#include <hip/hip_runtime.h>

typedef __attribute__((ext_vector_type(16))) _Float16 v16h;
typedef __attribute__((ext_vector_type(8)))  float    v8f;

#define B_      16
#define W_      128
#define L_      20
#define LL_     400
#define KP_     416   // K padded to 13 chunks of 32
#define NKC_    13
#define STRIDE_ 424   // f16 elements per LDS row (banking pad)

__global__ __launch_bounds__(256)
void coocc_wmma_kernel(const int* __restrict__ nodes,
                       const float* __restrict__ masks,
                       const float* __restrict__ kern,
                       float* __restrict__ out)
{
    __shared__ __align__(16) _Float16 Ebuf[W_ * STRIDE_];
    __shared__ __align__(16) _Float16 Gbuf[W_ * STRIDE_];
    __shared__ float validS[W_];
    __shared__ float Ks[LL_];

    const int b    = blockIdx.x;
    const int tid  = threadIdx.x;
    const int lane = tid & 31;
    const int wave = tid >> 5;

    // ---- Phase 1: zero E/G, stage kernel matrix ----
    {
        unsigned int* e32 = (unsigned int*)Ebuf;
        unsigned int* g32 = (unsigned int*)Gbuf;
        const int n32 = W_ * STRIDE_ / 2;
        for (int idx = tid; idx < n32; idx += 256) { e32[idx] = 0u; g32[idx] = 0u; }
        for (int idx = tid; idx < LL_; idx += 256) Ks[idx] = kern[idx];
    }
    __syncthreads();

    // ---- Phase 2: build E (one-hot indicators), valid sums, then G ----
    {
        const int r    = tid & 127;
        const int half = tid >> 7;   // 0 or 1: owns j/i range [half*10, half*10+10)
        const int*   arow = nodes + ((size_t)b * W_ + r) * L_;
        const float* mrow = masks + ((size_t)b * W_ + r) * L_;

        for (int i = half * 10; i < half * 10 + 10; ++i) {
            int   v = arow[i];
            float m = mrow[i];
            if (m > 0.f) Ebuf[r * STRIDE_ + v * L_ + i] = (_Float16)1.0f;
        }
        if (half == 0) {
            float s = 0.f;
            for (int i = 0; i < L_; ++i) s += mrow[i];
            validS[r] = s;
        }
        __syncthreads();

        // G[r][v*20+j] = sum over i with a[r,i]==v of K[i,j]; thread owns j-half
        const int jlo = half * 10;
        for (int i = 0; i < L_; ++i) {
            int   v = arow[i];
            float m = mrow[i];
            if (m > 0.f) {
                const float* kr = &Ks[i * L_];
                _Float16* gr = &Gbuf[r * STRIDE_ + v * L_ + jlo];
                for (int j = 0; j < 10; ++j)
                    gr[j] = (_Float16)((float)gr[j] + kr[jlo + j]);
            }
        }
    }
    __syncthreads();

    // ---- Phase 3: GEMM  C[r0..r0+15][0..127] = G-strip x E^T ----
    const int r0   = wave * 16;
    const int ln15 = lane & 15;
    const int lhi  = lane >> 4;

    v8f acc[8] = {};

    const _Float16* Abase = &Gbuf[(r0 + ln15) * STRIDE_ + lhi * 8];

    for (int kc = 0; kc < NKC_; ++kc) {
        const int k0 = kc * 32;
        union { uint4 u[2]; v16h h; } fa;
        fa.u[0] = *(const uint4*)(Abase + k0);        // K = k0+koff+0..7
        fa.u[1] = *(const uint4*)(Abase + k0 + 16);   // K = k0+16+koff+0..7
#pragma unroll
        for (int st = 0; st < 8; ++st) {
            const int scol = st * 16 + ln15;          // B col N = lane&15
            union { uint4 u[2]; v16h h; } fb;
            const _Float16* Bbase = &Ebuf[scol * STRIDE_ + k0 + lhi * 8];
            fb.u[0] = *(const uint4*)(Bbase);         // K = k0+koff+0..7
            fb.u[1] = *(const uint4*)(Bbase + 16);    // K = k0+16+koff+0..7
            acc[st] = __builtin_amdgcn_wmma_f32_16x16x32_f16(
                false, fa.h, false, fb.h, (short)0, acc[st], false, false);
        }
    }

    // ---- Phase 4: normalized store (f32 C/D layout) ----
    const int mbase = r0 + lhi * 8;
#pragma unroll
    for (int st = 0; st < 8; ++st) {
        const int s  = st * 16 + ln15;
        const float vs = validS[s];
#pragma unroll
        for (int d = 0; d < 8; ++d) {
            const int m = mbase + d;
            const float norm = validS[m] * vs + 1e-8f;
            out[((size_t)b * W_ + m) * W_ + s] = acc[st][d] / norm;
        }
    }
}

extern "C" void kernel_launch(void* const* d_in, const int* in_sizes, int n_in,
                              void* d_out, int out_size, void* d_ws, size_t ws_size,
                              hipStream_t stream) {
    const int*   nodes = (const int*)d_in[0];
    const float* masks = (const float*)d_in[1];
    const float* kern  = (const float*)d_in[2];
    float*       out   = (float*)d_out;
    coocc_wmma_kernel<<<dim3(B_), dim3(256), 0, stream>>>(nodes, masks, kern, out);
}
